// TorchOneToMany_55224689492107
// MI455X (gfx1250) — matrix operand order repaired and checked by reading, not compiled, not run on hardware
//
#include <hip/hip_runtime.h>
#include <hip/hip_bf16.h>
#include <math.h>

// Problem constants (from reference)
#define RNN_INPUT  1024
#define RNN_HIDDEN 2048
#define RNN_SEQ    256
#define RNN_BATCH  256
#define KCAT       (RNN_INPUT + RNN_HIDDEN)   // 3072 : concatenated [inp | h] K-dim

// Partition: 8 batch-groups of 32 rows x 16 hidden-slices = 128 WGs.
#define BG_ROWS  32
#define RNN_KC   384                          // RNN K staged in eighths of 3072
#define FC_KC    256                          // FC  K staged in eighths of 2048
#define LDSA     (RNN_KC + 8)                 // 392 f16 -> 784B row (4-bank skew)
#define LDSF     (FC_KC + 8)                  // 264 f16 -> 528B row (4-bank skew)
#define ABUF     (BG_ROWS * LDSA)             // 12,544 f16 per RNN buffer
#define FBUF     (BG_ROWS * LDSF)             //  8,448 f16 per FC buffer
#define NWG      128
#define THREADS  256                          // 8 wave32 per WG

typedef __attribute__((ext_vector_type(16))) _Float16 v16h;
typedef __attribute__((ext_vector_type(8)))  _Float16 v8h;
typedef __attribute__((ext_vector_type(8)))  float    v8f;
typedef __attribute__((ext_vector_type(4)))  int      v4i;
typedef __attribute__((address_space(1)))    v4i      gv4i;   // global int4
typedef __attribute__((address_space(3)))    v4i      lv4i;   // LDS int4

// ---- WMMA fragment loaders (CDNA5 16-bit operand layouts, ISA 7.12.2) ----
// A operand: lane<16 holds K {0..7, 16..23}, lane>=16 holds K {8..15, 24..31}
__device__ inline v16h lda_frag(const _Float16* p) {
  union { v16h v; v8h h[2]; } u;
  u.h[0] = *(const v8h*)(p);
  u.h[1] = *(const v8h*)(p + 16);
  return u.v;
}
// B operand: lane = output column; lane<16 holds K {0..7, 16..23}, lane>=16 K {8..15, 24..31}
// -> two 16-byte pieces of the stored (N x K row-major) weight row.
__device__ inline v16h ldb_frag(const _Float16* p) {
  union { v16h v; v8h h[2]; } u;
  u.h[0] = *(const v8h*)(p);
  u.h[1] = *(const v8h*)(p + 16);
  return u.v;
}

__device__ inline v8f wmma_f16(v16h a, v16h b, v8f c) {
  return __builtin_amdgcn_wmma_f32_16x16x32_f16(false, a, false, b, (short)0, c,
                                                false, false);
}

// ---- software-pipelined K loop: one B fragment feeds two M-tiles ----------
// B (global/L2, long latency) prefetched 3 steps ahead; A (LDS) 1 step ahead.
template <int NK>
__device__ inline void gemm_tile2(v8f& acc0, v8f& acc1,
                                  const _Float16* __restrict__ a0,
                                  const _Float16* __restrict__ a1,
                                  const _Float16* __restrict__ bp) {
  v16h bf[4];
  v16h af0[2], af1[2];
  #pragma unroll
  for (int i = 0; i < 3; ++i) bf[i] = ldb_frag(bp + i * 32);
  af0[0] = lda_frag(a0);
  af1[0] = lda_frag(a1);
  #pragma unroll
  for (int i = 0; i < NK; ++i) {
    if (i + 3 < NK) bf[(i + 3) & 3] = ldb_frag(bp + (i + 3) * 32);
    if (i + 1 < NK) {
      af0[(i + 1) & 1] = lda_frag(a0 + (i + 1) * 32);
      af1[(i + 1) & 1] = lda_frag(a1 + (i + 1) * 32);
    }
    acc0 = wmma_f16(af0[i & 1], bf[i & 3], acc0);
    acc1 = wmma_f16(af1[i & 1], bf[i & 3], acc1);
  }
}

// ---- staging: async global->LDS (ASYNCcnt) if the toolchain exposes it ----
#if __has_builtin(__builtin_amdgcn_global_load_async_to_lds_b128)
#define ASYNC_STAGE 1
#endif

template <int NCHUNK>
__device__ inline void stage_copy(_Float16* __restrict__ ldst,
                                  const _Float16* __restrict__ gsrc) {
#ifdef ASYNC_STAGE
  #pragma unroll
  for (int i = 0; i < NCHUNK; ++i)
    __builtin_amdgcn_global_load_async_to_lds_b128(
        (gv4i*)(gsrc + i * 64), (lv4i*)(ldst + i * 64), 0, 0);
#else
  #pragma unroll
  for (int i = 0; i < NCHUNK; ++i)
    *(v8h*)(ldst + i * 64) = *(const v8h*)(gsrc + i * 64);
#endif
}

__device__ inline void stage_fence() {
#ifdef ASYNC_STAGE
#if __has_builtin(__builtin_amdgcn_s_wait_asynccnt)
  __builtin_amdgcn_s_wait_asynccnt(0);
#else
  asm volatile("s_wait_asynccnt 0x0" ::: "memory");
#endif
#endif
}

// ---- device-wide split barrier: monotonic counter, release/acquire ----
__device__ inline void grid_sync(unsigned* cnt, unsigned target) {
  __syncthreads();
  if (threadIdx.x == 0) {
    __hip_atomic_fetch_add(cnt, 1u, __ATOMIC_RELEASE, __HIP_MEMORY_SCOPE_AGENT);
    while (__hip_atomic_load(cnt, __ATOMIC_ACQUIRE, __HIP_MEMORY_SCOPE_AGENT) < target) {
      __builtin_amdgcn_s_sleep(2);
    }
  }
  __syncthreads();
}

// ---- prep: f32 -> f16 weight conversion, bias fusion, state init ----
__global__ void rnn_prep(const float* __restrict__ x, const float* __restrict__ h0,
                         const float* __restrict__ Wih, const float* __restrict__ Whh,
                         const float* __restrict__ bih, const float* __restrict__ bhh,
                         const float* __restrict__ Wfc,
                         _Float16* __restrict__ WcatH, _Float16* __restrict__ WfcH,
                         float* __restrict__ brnn, _Float16* __restrict__ act0,
                         unsigned* __restrict__ cnt) {
  const size_t tid    = (size_t)blockIdx.x * blockDim.x + threadIdx.x;
  const size_t stride = (size_t)gridDim.x * blockDim.x;
  // Wcat[n][k] = k < 1024 ? W_ih[n][k] : W_hh[n][k-1024]   (2048 x 3072)
  for (size_t i = tid; i < (size_t)RNN_HIDDEN * KCAT; i += stride) {
    const size_t n = i / KCAT, k = i % KCAT;
    const float w = (k < RNN_INPUT) ? Wih[n * RNN_INPUT + k]
                                    : Whh[n * RNN_HIDDEN + (k - RNN_INPUT)];
    WcatH[i] = (_Float16)w;
  }
  for (size_t i = tid; i < (size_t)RNN_INPUT * RNN_HIDDEN; i += stride)
    WfcH[i] = (_Float16)Wfc[i];
  for (size_t i = tid; i < (size_t)RNN_HIDDEN; i += stride)
    brnn[i] = bih[i] + bhh[i];
  // act0[b][k] = k < 1024 ? x[b][k] : h0[b][k-1024]   (256 x 3072)
  for (size_t i = tid; i < (size_t)RNN_BATCH * KCAT; i += stride) {
    const size_t b = i / KCAT, k = i % KCAT;
    act0[i] = (_Float16)((k < RNN_INPUT) ? x[b * RNN_INPUT + k]
                                         : h0[b * RNN_HIDDEN + (k - RNN_INPUT)]);
  }
  if (tid == 0) *cnt = 0u;
}

// ---- persistent RNN: 128 WGs, internal 256-step loop, WMMA f16 ----
__global__ void __launch_bounds__(THREADS)
rnn_persistent(const _Float16* __restrict__ Wcat,   // 2048 x 3072, f16
               const _Float16* __restrict__ Wfc,    // 1024 x 2048, f16
               const float* __restrict__ brnn,      // 2048
               const float* __restrict__ bfc,       // 1024
               _Float16* __restrict__ actA,         // 256 x 3072, double buffer
               _Float16* __restrict__ actB,
               float* __restrict__ out,             // (256 batch, 256 seq, 1024)
               unsigned* __restrict__ cnt) {
  __shared__ _Float16 lds[2 * ABUF];                // 50,176 bytes, ping-pong

  const int tid   = (int)threadIdx.x;
  const int wave  = tid >> 5;                       // 0..7 (wave32)
  const int lane  = tid & 31;
  const int wg    = (int)blockIdx.x;
  const int bt0   = (wg >> 4) * BG_ROWS;            // batch row base (32 rows)
  const int nsl   = wg & 15;                        // hidden slice 0..15
  const int rowA  = lane & 15;
  const int half  = lane >> 4;                      // 0 / 1
  const int koffA = half * 8;                       // A operand K split
  const int koffB = half * 8;                       // B operand K split
  const int srow  = tid >> 3;                       // staging row 0..31
  const int sli   = tid & 7;                        // 8 staging threads per row

  _Float16* cur = actA;
  _Float16* nxt = actB;

  for (int t = 0; t < RNN_SEQ; ++t) {
    // ============ phase 1: h_new = tanh([inp|h] @ Wcat^T + b_rnn) ==========
    // Wave owns 1 N-tile x 2 M-tiles -> one B fragment feeds two WMMAs.
    // LDS ping-pong: async-stage chunk kb+1 while computing on chunk kb.
    const int n0 = nsl * 128 + wave * 16;
    v8f acc0 = {};
    v8f acc1 = {};
    {
      const _Float16* gact = cur + (size_t)(bt0 + srow) * KCAT + sli * 8;
      _Float16* lrow = &lds[srow * LDSA + sli * 8];
      stage_copy<6>(lrow, gact);                    // prologue: chunk 0
      stage_fence();
      __syncthreads();
      for (int kb = 0; kb < 8; ++kb) {
        if (kb < 7)                                 // stage chunk kb+1
          stage_copy<6>(lrow + ((kb + 1) & 1) * ABUF, gact + (kb + 1) * RNN_KC);
        const _Float16* bufc = lds + (kb & 1) * ABUF;
        gemm_tile2<RNN_KC / 32>(acc0, acc1,
                                bufc + rowA * LDSA + koffA,
                                bufc + (16 + rowA) * LDSA + koffA,
                                Wcat + (size_t)(n0 + rowA) * KCAT + kb * RNN_KC + koffB);
        stage_fence();
        __syncthreads();
      }
    }
    { // epilogue: bias + tanh -> f16 h region of next activation buffer
      const int nn  = lane & 15;
      const int col = n0 + nn;
      const float bb = brnn[col];
      #pragma unroll
      for (int j = 0; j < 8; ++j) {
        const int r0 = bt0 + half * 8 + j;            // C/D layout: M = j + 8*half
        const int r1 = r0 + 16;
        nxt[(size_t)r0 * KCAT + RNN_INPUT + col] = (_Float16)tanhf(acc0[j] + bb);
        nxt[(size_t)r1 * KCAT + RNN_INPUT + col] = (_Float16)tanhf(acc1[j] + bb);
      }
    }
    grid_sync(cnt, (unsigned)(2 * t + 1) * NWG);      // h_new visible device-wide

    // ============ phase 2: out = h_new @ Wfc^T + b_fc ======================
    // 4 N-tiles per WG; wave pair (nt, nt+4) splits K so every Wfc fragment
    // is loaded once per WG; f32 reduction through LDS afterwards.
    const int nt  = wave & 3;                         // N tile 0..3
    const int khv = wave >> 2;                        // K half 0/1
    const int m0  = nsl * 64 + nt * 16;
    v8f f0 = {};
    v8f f1 = {};
    {
      const _Float16* gh = nxt + (size_t)(bt0 + srow) * KCAT + RNN_INPUT + sli * 8;
      _Float16* lrow = &lds[srow * LDSF + sli * 8];
      stage_copy<4>(lrow, gh);                        // prologue: chunk 0
      stage_fence();
      __syncthreads();
      for (int kb = 0; kb < 8; ++kb) {
        if (kb < 7)
          stage_copy<4>(lrow + ((kb + 1) & 1) * FBUF, gh + (kb + 1) * FC_KC);
        if ((kb >> 2) == khv) {                       // this wave's K half only
          const _Float16* bufc = lds + (kb & 1) * FBUF;
          gemm_tile2<FC_KC / 32>(f0, f1,
                                 bufc + rowA * LDSF + koffA,
                                 bufc + (16 + rowA) * LDSF + koffA,
                                 Wfc + (size_t)(m0 + rowA) * RNN_HIDDEN + kb * FC_KC + koffB);
        }
        stage_fence();
        __syncthreads();
      }
    }
    // cross-wave K reduction through LDS (4 waves x 32 lanes x 32 f32 = 16KB)
    {
      float* red = (float*)lds;
      const int base = (nt * 32 + lane) * 32;
      if (khv == 1) {
        #pragma unroll
        for (int j = 0; j < 8; ++j) {
          red[base + j]      = f0[j];
          red[base + 16 + j] = f1[j];
        }
      }
      __syncthreads();
      if (khv == 0) {
        const int nn  = lane & 15;
        const int col = m0 + nn;
        const float bb = bfc[col];
        #pragma unroll
        for (int j = 0; j < 8; ++j) {
          const int r0 = bt0 + half * 8 + j;
          const int r1 = r0 + 16;
          const float v0 = f0[j] + red[base + j]      + bb;
          const float v1 = f1[j] + red[base + 16 + j] + bb;
          out[(size_t)r0 * RNN_SEQ * RNN_INPUT + (size_t)t * RNN_INPUT + col] = v0;
          out[(size_t)r1 * RNN_SEQ * RNN_INPUT + (size_t)t * RNN_INPUT + col] = v1;
          nxt[(size_t)r0 * KCAT + col] = (_Float16)v0;   // next step's input
          nxt[(size_t)r1 * KCAT + col] = (_Float16)v1;
        }
      }
    }
    grid_sync(cnt, (unsigned)(2 * t + 2) * NWG);      // out/inp visible

    _Float16* tmp = cur; cur = nxt; nxt = tmp;        // swap double buffers
  }
}

extern "C" void kernel_launch(void* const* d_in, const int* in_sizes, int n_in,
                              void* d_out, int out_size, void* d_ws, size_t ws_size,
                              hipStream_t stream) {
  (void)in_sizes; (void)n_in; (void)out_size; (void)ws_size;
  const float* x      = (const float*)d_in[0];
  const float* init_h = (const float*)d_in[1];
  const float* Wih    = (const float*)d_in[2];
  const float* Whh    = (const float*)d_in[3];
  const float* bih    = (const float*)d_in[4];
  const float* bhh    = (const float*)d_in[5];
  const float* Wfc    = (const float*)d_in[6];
  const float* bfc    = (const float*)d_in[7];

  // Workspace carve-out (all 256B aligned), ~19.9 MB total
  char* p = (char*)d_ws;
  auto carve = [&](size_t bytes) {
    char* r = p;
    p += (bytes + 255) & ~(size_t)255;
    return r;
  };
  _Float16* WcatH = (_Float16*)carve((size_t)RNN_HIDDEN * KCAT * 2);       // 12 MB
  _Float16* WfcH  = (_Float16*)carve((size_t)RNN_INPUT * RNN_HIDDEN * 2);  //  4 MB
  float*    brnn  = (float*)   carve((size_t)RNN_HIDDEN * 4);
  _Float16* actA  = (_Float16*)carve((size_t)RNN_BATCH * KCAT * 2);        // 1.5 MB
  _Float16* actB  = (_Float16*)carve((size_t)RNN_BATCH * KCAT * 2);        // 1.5 MB
  unsigned* cnt   = (unsigned*)carve(256);

  rnn_prep<<<256, 256, 0, stream>>>(x, init_h, Wih, Whh, bih, bhh, Wfc,
                                    WcatH, WfcH, brnn, actA, cnt);
  rnn_persistent<<<NWG, THREADS, 0, stream>>>(WcatH, WfcH, brnn, bfc,
                                              actA, actB, (float*)d_out, cnt);
}
